// SemPosBlock_13494787244855
// MI455X (gfx1250) — matrix operand order repaired and checked by reading, not compiled, not run on hardware
//
#include <hip/hip_runtime.h>

// ---------------------------------------------------------------------------
// SemPosBlock attention for MI455X (gfx1250, wave32, WMMA).
//
//   T=256 frames, S=128 persons, F=1024 features.
//   theta = x@Wa+ba, phi = x@Wb+bb, feats = x@Wg+bg          (206 GFLOP)
//   per frame: out = softmax(theta phi^T) feats              ( 17 GFLOP)
//
// Projections: split-precision bf16x3 via V_WMMA_F32_16X16X32_BF16
//   (x = hi + lo bf16 planes; acc += Ah*Bh + Ah*Bl + Al*Bh, fp32 accumulate;
//    ~2^-17 relative error, 2.7x the FLOP/instr of the f32 WMMA path).
// Attention: fp32 via V_WMMA_F32_16X16X4_F32 (only 8% of total FLOPs);
//   tile staging uses GLOBAL_LOAD_ASYNC_TO_LDS_B128 (ASYNCcnt-tracked DMA,
//   no VGPR round-trip).
// Workspace: 3*T*S*F*4 = 402 MB in d_ws for theta/phi/feats.
// ---------------------------------------------------------------------------

typedef __attribute__((ext_vector_type(2))) float v2f;
typedef __attribute__((ext_vector_type(8))) float v8f;
typedef __attribute__((ext_vector_type(8))) __bf16 v8bf;
typedef __attribute__((ext_vector_type(16))) __bf16 v16bf;

#define T_FRAMES 256
#define S_PERS   128
#define FDIM     1024

// ---------------- float <-> bf16 split helpers -----------------------------
__device__ __forceinline__ unsigned short f2bf(float f) {
  unsigned int u = __float_as_uint(f);
  u += 0x7FFFu + ((u >> 16) & 1u);  // round-to-nearest-even
  return (unsigned short)(u >> 16);
}
__device__ __forceinline__ float bf2f(unsigned short h) {
  return __uint_as_float(((unsigned int)h) << 16);
}

// ---------------- async global -> LDS copy (16 B per lane) -----------------
// LDS dest address from the vdst VGPR; GV mode global addr from VGPR
// pair. Tracked on ASYNCcnt; waited with s_wait_asynccnt before the barrier.
__device__ __forceinline__ void async_g2l_b128(void* lds, const void* gptr) {
  unsigned loff = (unsigned)(size_t)lds;  // LDS aperture keeps offset in [31:0]
  asm volatile("global_load_async_to_lds_b128 %0, %1, off"
               :
               : "v"(loff), "v"(gptr)
               : "memory");
}
__device__ __forceinline__ void wait_async0() {
  asm volatile("s_wait_asynccnt 0x0" ::: "memory");
}

// ---------------- f32 WMMA (16x16x4) operand fetch -------------------------
// Row-major [row][k] LDS tile; matches f32 A (16x4) layout, and B (4x16) when
// rows are the N index of a [n][k]-stored tile:
//   VGPR0: lanes 0-15 -> K=kk,   lanes 16-31 -> K=kk+2
//   VGPR1: lanes 0-15 -> K=kk+1, lanes 16-31 -> K=kk+3
__device__ __forceinline__ v2f ld_mk(const float* base, int pitch, int row0,
                                     int kk, int lane) {
  int r = row0 + (lane & 15);
  int k = kk + ((lane >> 4) << 1);
  float2 t = *(const float2*)(base + r * pitch + k);
  v2f v;
  v.x = t.x;
  v.y = t.y;
  return v;
}

// B-operand fetch from a row-major [k][n] LDS tile (f32 path).
__device__ __forceinline__ v2f ld_b_rm(const float* base, int pitch, int col0,
                                       int kk, int lane) {
  int c = col0 + (lane & 15);
  int k = kk + ((lane >> 4) << 1);
  v2f v;
  v.x = base[(k + 0) * pitch + c];
  v.y = base[(k + 1) * pitch + c];
  return v;
}

__device__ __forceinline__ v8f wmma_f32(v2f a, v2f b, v8f c) {
  return __builtin_amdgcn_wmma_f32_16x16x4_f32(false, a, false, b, (short)0, c,
                                               false, false);
}

// ---------------- bf16 WMMA (16x16x32) operand fetch -----------------------
// A 16x32 (ISA layout): lanes 0-15 hold K{0..7,16..23}, lanes 16-31 hold
// K{8..15,24..31}. Tile stored row-major [m][K=32], pitch 40 halves (80 B:
// 16B-aligned, odd multiple of 16B -> conflict-free b128 reads).
__device__ __forceinline__ v16bf ld_a_bf(const unsigned short* base, int pitch,
                                         int row0, int lane) {
  int r = row0 + (lane & 15);
  int klo = (lane >> 4) << 3;  // 0 | 8
  const unsigned short* p = base + r * pitch + klo;
  v8bf lo = *(const v8bf*)(p);       // K = klo .. klo+7
  v8bf hi = *(const v8bf*)(p + 16);  // K = klo+16 .. klo+23
  return __builtin_shufflevector(lo, hi, 0, 1, 2, 3, 4, 5, 6, 7, 8, 9, 10, 11,
                                 12, 13, 14, 15);
}

// B 32x16 (ISA layout): lanes 0-15 hold K{0..7,16..23} of col=lane, lanes 16-31
// hold K{8..15,24..31}. W tile stored transposed [n][K=32], pitch 40 halves.
__device__ __forceinline__ v16bf ld_b_bf(const unsigned short* base, int pitch,
                                         int col0, int lane) {
  int c = col0 + (lane & 15);
  int k0 = (lane >> 4) << 3;  // 0 | 8
  const unsigned short* p = base + c * pitch + k0;
  v8bf lo = *(const v8bf*)(p);      // K = k0 .. k0+7
  v8bf hi = *(const v8bf*)(p + 16);  // K = k0+16 .. k0+23
  return __builtin_shufflevector(lo, hi, 0, 1, 2, 3, 4, 5, 6, 7, 8, 9, 10, 11,
                                 12, 13, 14, 15);
}

__device__ __forceinline__ v8f wmma_bf16(v16bf a, v16bf b, v8f c) {
  return __builtin_amdgcn_wmma_f32_16x16x32_bf16(false, a, false, b, (short)0,
                                                 c, false, false);
}

// ---------------------------------------------------------------------------
// Kernel 1: three projection GEMMs  C[32768,1024] = X @ W + b  -> d_ws
// grid = (8 n-blocks, 256 m-blocks, 3 projections), block = 256 (8 waves).
// Block tile 128x128, Kc = 32 (= one bf16 WMMA K). Wave tile 32x64.
// Split-precision bf16x3: per 16x16 tile per K-chunk, 3 x wmma_bf16.
// (Register staging kept here: tiles are converted f32 -> hi/lo bf16 on the
//  way to LDS, which an async DMA copy cannot do.)
// ---------------------------------------------------------------------------
__global__ __launch_bounds__(256) void proj_gemm_kernel(
    const float* __restrict__ X, const float* __restrict__ Wa,
    const float* __restrict__ ba, const float* __restrict__ Wb,
    const float* __restrict__ bb, const float* __restrict__ Wg,
    const float* __restrict__ bg, float* __restrict__ ws) {
  __shared__ __align__(16) unsigned short Xhi[128][40];  // X hi plane, [m][k]
  __shared__ __align__(16) unsigned short Xlo[128][40];  // X lo plane
  __shared__ __align__(16) unsigned short Whi[128][40];  // W^T [n][k], hi
  __shared__ __align__(16) unsigned short Wlo[128][40];  // W^T [n][k], lo

  const int tid = threadIdx.x;
  const int lane = tid & 31;
  const int wave = tid >> 5;
  const int wm = (wave >> 1) * 32;  // wave M offset in block tile
  const int wn = (wave & 1) * 64;   // wave N offset in block tile
  const int n_base = blockIdx.x * 128;
  const int m_base = blockIdx.y * 128;
  const int proj = blockIdx.z;

  const float* Wp = (proj == 0) ? Wa : ((proj == 1) ? Wb : Wg);
  const float* bp = (proj == 0) ? ba : ((proj == 1) ? bb : bg);
  float* outp = ws + (size_t)proj * (size_t)T_FRAMES * S_PERS * FDIM;

  v8f acc[2][4];
#pragma unroll
  for (int mi = 0; mi < 2; ++mi)
#pragma unroll
    for (int ni = 0; ni < 4; ++ni) acc[mi][ni] = (v8f){};

  for (int k0 = 0; k0 < FDIM; k0 += 32) {
    __syncthreads();
    // X tile: 128x32 f32, coalesced float4 loads, split to hi/lo bf16 planes.
#pragma unroll
    for (int i = 0; i < 4; ++i) {
      int f4 = tid + i * 256;  // float4 index, 0..1023
      int row = f4 >> 3;       // 8 float4 per row
      int c = (f4 & 7) << 2;
      float4 v = *(const float4*)(X + (size_t)(m_base + row) * FDIM + k0 + c);
      unsigned short h0 = f2bf(v.x), h1 = f2bf(v.y), h2 = f2bf(v.z),
                     h3 = f2bf(v.w);
      unsigned short l0 = f2bf(v.x - bf2f(h0)), l1 = f2bf(v.y - bf2f(h1)),
                     l2 = f2bf(v.z - bf2f(h2)), l3 = f2bf(v.w - bf2f(h3));
      uint2 ph, pl;
      ph.x = (unsigned)h0 | ((unsigned)h1 << 16);
      ph.y = (unsigned)h2 | ((unsigned)h3 << 16);
      pl.x = (unsigned)l0 | ((unsigned)l1 << 16);
      pl.y = (unsigned)l2 | ((unsigned)l3 << 16);
      *(uint2*)&Xhi[row][c] = ph;
      *(uint2*)&Xlo[row][c] = pl;
    }
    // W tile: 32x128 f32, coalesced loads, transposed split store -> [n][k].
#pragma unroll
    for (int i = 0; i < 4; ++i) {
      int f4 = tid + i * 256;
      int kr = f4 >> 5;  // 32 float4 per k-row
      int n = (f4 & 31) << 2;
      float4 v = *(const float4*)(Wp + (size_t)(k0 + kr) * FDIM + n_base + n);
      float fv[4] = {v.x, v.y, v.z, v.w};
#pragma unroll
      for (int j = 0; j < 4; ++j) {
        unsigned short h = f2bf(fv[j]);
        unsigned short l = f2bf(fv[j] - bf2f(h));
        Whi[n + j][kr] = h;
        Wlo[n + j][kr] = l;
      }
    }
    __syncthreads();

    // One K=32 chunk: 2 A tiles x 4 B tiles x 3 split products = 24 WMMAs.
    v16bf a0h = ld_a_bf(&Xhi[0][0], 40, wm, lane);
    v16bf a0l = ld_a_bf(&Xlo[0][0], 40, wm, lane);
    v16bf a1h = ld_a_bf(&Xhi[0][0], 40, wm + 16, lane);
    v16bf a1l = ld_a_bf(&Xlo[0][0], 40, wm + 16, lane);
#pragma unroll
    for (int ni = 0; ni < 4; ++ni) {
      v16bf bh = ld_b_bf(&Whi[0][0], 40, wn + ni * 16, lane);
      v16bf bl = ld_b_bf(&Wlo[0][0], 40, wn + ni * 16, lane);
      acc[0][ni] = wmma_bf16(a0h, bh, acc[0][ni]);
      acc[0][ni] = wmma_bf16(a0h, bl, acc[0][ni]);
      acc[0][ni] = wmma_bf16(a0l, bh, acc[0][ni]);
      acc[1][ni] = wmma_bf16(a1h, bh, acc[1][ni]);
      acc[1][ni] = wmma_bf16(a1h, bl, acc[1][ni]);
      acc[1][ni] = wmma_bf16(a1l, bh, acc[1][ni]);
    }
  }

  // Store C + bias. C layout: VGPR r -> (M=r | M=r+8), N = lane&15.
#pragma unroll
  for (int mi = 0; mi < 2; ++mi) {
    int rbase = m_base + wm + mi * 16 + ((lane >> 4) << 3);
#pragma unroll
    for (int ni = 0; ni < 4; ++ni) {
      int colg = n_base + wn + ni * 16 + (lane & 15);
      float bias = bp[colg];
#pragma unroll
      for (int r = 0; r < 8; ++r)
        outp[(size_t)(rbase + r) * FDIM + colg] = acc[mi][ni][r] + bias;
    }
  }
}

// ---------------------------------------------------------------------------
// Kernel 2: per-frame attention. One block (8 waves) per frame t.
//   scores = theta_t phi_t^T  (K=1024, f32 WMMA)  -> LDS
//   softmax rows               (shfl + expf)
//   out    = scores feats_t    (K=128,  f32 WMMA)
// Tile staging via async global->LDS DMA (ASYNCcnt).
// ---------------------------------------------------------------------------
__global__ __launch_bounds__(256) void attn_kernel(const float* __restrict__ ws,
                                                   float* __restrict__ out) {
  __shared__ __align__(16) float sc[128][132];  // scores / softmax weights
  union TileU {
    struct {
      float th[128][36];
      float ph[128][36];
    } p;                // phase-1 theta/phi K-tiles
    float fc[128][68];  // phase-3 feats N-chunk (128x64)
  };
  __shared__ __align__(16) TileU u;

  const int tid = threadIdx.x;
  const int lane = tid & 31;
  const int wave = tid >> 5;
  const int t = blockIdx.x;
  const int m0 = wave * 16;  // this wave owns query rows m0..m0+15
  const size_t TSF = (size_t)T_FRAMES * S_PERS * FDIM;
  const float* theta = ws;
  const float* phi = ws + TSF;
  const float* feats = ws + 2 * TSF;
  const size_t rowbase = (size_t)t * S_PERS;

  // ---- Phase 1: scores[128][128] = theta @ phi^T --------------------------
  v8f acc[8];
#pragma unroll
  for (int nt = 0; nt < 8; ++nt) acc[nt] = (v8f){};

  for (int k0 = 0; k0 < FDIM; k0 += 32) {
    __syncthreads();  // previous tile fully consumed before DMA overwrites
#pragma unroll
    for (int i = 0; i < 4; ++i) {
      int f4 = tid + i * 256;
      int row = f4 >> 3;
      int c = (f4 & 7) << 2;
      async_g2l_b128(&u.p.th[row][c],
                     theta + (rowbase + row) * FDIM + k0 + c);
      async_g2l_b128(&u.p.ph[row][c], phi + (rowbase + row) * FDIM + k0 + c);
    }
    wait_async0();
    __syncthreads();

#pragma unroll
    for (int kk = 0; kk < 32; kk += 4) {
      v2f a = ld_mk(&u.p.th[0][0], 36, m0, kk, lane);
#pragma unroll
      for (int nt = 0; nt < 8; ++nt) {
        // phi stored [j][k] row-major == B[k][j] = phi^T, exactly B layout.
        v2f b = ld_mk(&u.p.ph[0][0], 36, nt * 16, kk, lane);
        acc[nt] = wmma_f32(a, b, acc[nt]);
      }
    }
  }

  // Spill scores to LDS (each wave writes only its own 16 rows).
  {
    int rbase = m0 + ((lane >> 4) << 3);
#pragma unroll
    for (int nt = 0; nt < 8; ++nt) {
      int col = nt * 16 + (lane & 15);
#pragma unroll
      for (int r = 0; r < 8; ++r) sc[rbase + r][col] = acc[nt][r];
    }
  }
  __syncthreads();

  // ---- Phase 2: row softmax (wave handles its 16 rows, 4 cols/lane) -------
  for (int r = 0; r < 16; ++r) {
    float* rowp = &sc[m0 + r][0];
    float x0 = rowp[lane + 0];
    float x1 = rowp[lane + 32];
    float x2 = rowp[lane + 64];
    float x3 = rowp[lane + 96];
    float mx = fmaxf(fmaxf(x0, x1), fmaxf(x2, x3));
#pragma unroll
    for (int off = 16; off > 0; off >>= 1)
      mx = fmaxf(mx, __shfl_xor(mx, off, 32));
    float e0 = __expf(x0 - mx);
    float e1 = __expf(x1 - mx);
    float e2 = __expf(x2 - mx);
    float e3 = __expf(x3 - mx);
    float s = e0 + e1 + e2 + e3;
#pragma unroll
    for (int off = 16; off > 0; off >>= 1) s += __shfl_xor(s, off, 32);
    float inv = 1.0f / s;
    rowp[lane + 0] = e0 * inv;
    rowp[lane + 32] = e1 * inv;
    rowp[lane + 64] = e2 * inv;
    rowp[lane + 96] = e3 * inv;
  }
  __syncthreads();

  // ---- Phase 3: out = weights @ feats, N chunked by 64 --------------------
  for (int n0 = 0; n0 < FDIM; n0 += 64) {
    __syncthreads();
#pragma unroll
    for (int i = 0; i < 8; ++i) {
      int f4 = tid + i * 256;
      int row = f4 >> 4;  // 16 float4 per row (64 floats)
      int c = (f4 & 15) << 2;
      async_g2l_b128(&u.fc[row][c], feats + (rowbase + row) * FDIM + n0 + c);
    }
    wait_async0();
    __syncthreads();

    v8f o[4];
#pragma unroll
    for (int ni = 0; ni < 4; ++ni) o[ni] = (v8f){};

#pragma unroll 8
    for (int kk = 0; kk < 128; kk += 4) {
      v2f a = ld_mk(&sc[0][0], 132, m0, kk, lane);
#pragma unroll
      for (int ni = 0; ni < 4; ++ni) {
        v2f b = ld_b_rm(&u.fc[0][0], 68, ni * 16, kk, lane);
        o[ni] = wmma_f32(a, b, o[ni]);
      }
    }

    int rbase = t * S_PERS + m0 + ((lane >> 4) << 3);
#pragma unroll
    for (int ni = 0; ni < 4; ++ni) {
      int colg = n0 + ni * 16 + (lane & 15);
#pragma unroll
      for (int r = 0; r < 8; ++r)
        out[(size_t)(rbase + r) * FDIM + colg] = o[ni][r];
    }
  }
}

// ---------------------------------------------------------------------------
extern "C" void kernel_launch(void* const* d_in, const int* in_sizes, int n_in,
                              void* d_out, int out_size, void* d_ws,
                              size_t ws_size, hipStream_t stream) {
  (void)in_sizes;
  (void)n_in;
  (void)out_size;
  (void)ws_size;  // requires >= 3*T*S*F*4 = 402,653,184 bytes

  const float* X = (const float*)d_in[0];  // batch_data [T*S, F]
  // d_in[1] = xywh (unused by reference output)
  const float* Wa = (const float*)d_in[2];
  const float* ba = (const float*)d_in[3];
  const float* Wb = (const float*)d_in[4];
  const float* bb = (const float*)d_in[5];
  const float* Wg = (const float*)d_in[6];
  const float* bg = (const float*)d_in[7];
  // d_in[8]=OW, d_in[9]=OH, d_in[10]=num_person : unused by reference output
  float* out = (float*)d_out;
  float* ws = (float*)d_ws;

  dim3 g1(FDIM / 128, (T_FRAMES * S_PERS) / 128, 3);
  proj_gemm_kernel<<<g1, 256, 0, stream>>>(X, Wa, ba, Wb, bb, Wg, bg, ws);
  attn_kernel<<<dim3(T_FRAMES), 256, 0, stream>>>(ws, out);
}
